// CausalAttention_ov2_57518202028181
// MI455X (gfx1250) — matrix operand order repaired and checked by reading, not compiled, not run on hardware
//
#include <hip/hip_runtime.h>
#include <stdint.h>

#define SEQ 4096
#define DIM 1024

typedef __attribute__((ext_vector_type(16))) __bf16 v16bf;
typedef __attribute__((ext_vector_type(8)))  float  v8f;
typedef __attribute__((ext_vector_type(4)))  unsigned int u32x4;
typedef __attribute__((ext_vector_type(8)))  int i32x8;
typedef __attribute__((ext_vector_type(4)))  int i32x4;

union FragB16 { uint4 q[2]; unsigned short us[16]; v16bf v; };

// fp32 -> bf16 round-to-nearest-even (finite inputs)
__device__ inline unsigned short f2bf(float f) {
  union { float f; unsigned u; } x; x.f = f;
  unsigned u = x.u;
  u += 0x7fffu + ((u >> 16) & 1u);
  return (unsigned short)(u >> 16);
}

__device__ inline void load_frag_a(FragB16& a, const unsigned short* aRow,
                                   int kk, int kbA) {
  a.q[0] = *(const uint4*)(aRow + kk + kbA);        // K = kk+kbA .. +7
  a.q[1] = *(const uint4*)(aRow + kk + kbA + 16);   // K = kk+kbA+16 .. +23
}
__device__ inline void load_frag_b(FragB16& b, const unsigned short* bRow,
                                   int kk, int koB) {
  b.q[0] = *(const uint4*)(bRow + kk + koB);        // K = kk+koB .. +7
  b.q[1] = *(const uint4*)(bRow + kk + koB + 16);   // K = kk+koB+16 .. +23
}

#define WMMA_BF16(A, B, C) \
  __builtin_amdgcn_wmma_f32_16x16x32_bf16(false, (A).v, false, (B).v, (short)0, (C), false, false)

// ---------------- Kernel 1: convert fp32 inputs to bf16 in workspace -------
__global__ __launch_bounds__(256) void cvt_bf16_kernel(
    const float* __restrict__ x,  const float* __restrict__ wq,
    const float* __restrict__ wk, const float* __restrict__ wv,
    unsigned short* __restrict__ xb,  unsigned short* __restrict__ wqb,
    unsigned short* __restrict__ wkb, unsigned short* __restrict__ wvb) {
  size_t i = (size_t)blockIdx.x * 256 + threadIdx.x;
  const size_t NX = (size_t)SEQ * DIM;       // 4194304
  const size_t NW = (size_t)DIM * DIM;       // 1048576
  if (i < NX)                { xb [i]             = f2bf(x [i]); }
  else if (i < NX + NW)      { wqb[i - NX]        = f2bf(wq[i - NX]); }
  else if (i < NX + 2 * NW)  { wkb[i - NX - NW]   = f2bf(wk[i - NX - NW]); }
  else if (i < NX + 3 * NW)  { wvb[i - NX - 2*NW] = f2bf(wv[i - NX - 2*NW]); }
}

// ---------------- Kernel 2: QKV projection GEMM (bf16 WMMA, 32x32/wave) ----
// out[n,o] = sum_k x[n,k] * W[o,k]   (torch Linear x @ W^T)
// Q,K stored row-major bf16 [SEQ][DIM]; V stored transposed Vt[DIM][SEQ].
__global__ __launch_bounds__(256) void qkv_gemm_kernel(
    const unsigned short* __restrict__ xb,
    const unsigned short* __restrict__ wqb,
    const unsigned short* __restrict__ wkb,
    const unsigned short* __restrict__ wvb,
    unsigned short* __restrict__ qb,
    unsigned short* __restrict__ kbuf,
    unsigned short* __restrict__ vtb) {
  const int wave = threadIdx.x >> 5;
  const int lane = threadIdx.x & 31;
  const int tid  = blockIdx.x * 8 + wave;      // 3 * 128 * 32 = 12288 wave-tiles
  const int mat  = tid >> 12;                  // 0=Q 1=K 2=V
  const int rem  = tid & 4095;
  const int tm   = rem >> 5;                   // 32-row seq tile  [0,128)
  const int to   = rem & 31;                   // 32-col out tile  [0,32)
  const unsigned short* wsrc = (mat == 0) ? wqb : (mat == 1) ? wkb : wvb;

  const int n    = lane & 15;                  // A-row m / B-col n
  const int kbA  = (lane < 16) ? 0 : 8;
  const int koB  = (lane < 16) ? 0 : 8;
  const int half = (lane < 16) ? 0 : 8;

  const unsigned short* aR0 = xb   + (size_t)(tm * 32 +      n) * DIM;
  const unsigned short* aR1 = xb   + (size_t)(tm * 32 + 16 + n) * DIM;
  const unsigned short* bR0 = wsrc + (size_t)(to * 32 +      n) * DIM;
  const unsigned short* bR1 = wsrc + (size_t)(to * 32 + 16 + n) * DIM;

  v8f acc00 = {}, acc01 = {}, acc10 = {}, acc11 = {};
  for (int kk = 0; kk < DIM; kk += 32) {
    FragB16 a0, a1, b0, b1;
    load_frag_a(a0, aR0, kk, kbA);
    load_frag_a(a1, aR1, kk, kbA);
    load_frag_b(b0, bR0, kk, koB);
    load_frag_b(b1, bR1, kk, koB);
    __builtin_prefetch(bR0 + kk + 128, 0, 3);  // WGP-scope global_prefetch_b8
    acc00 = WMMA_BF16(a0, b0, acc00);
    acc01 = WMMA_BF16(a0, b1, acc01);
    acc10 = WMMA_BF16(a1, b0, acc10);
    acc11 = WMMA_BF16(a1, b1, acc11);
  }

#pragma unroll
  for (int mi = 0; mi < 2; ++mi) {
#pragma unroll
    for (int ni = 0; ni < 2; ++ni) {
      const v8f acc = (mi == 0) ? (ni == 0 ? acc00 : acc01)
                                : (ni == 0 ? acc10 : acc11);
#pragma unroll
      for (int rr = 0; rr < 8; ++rr) {
        const int M  = rr + half;
        const int gm = tm * 32 + mi * 16 + M;
        const int go = to * 32 + ni * 16 + n;
        const unsigned short h = f2bf(acc[rr]);
        if (mat == 0)      qb  [(size_t)gm * DIM + go] = h;
        else if (mat == 1) kbuf[(size_t)gm * DIM + go] = h;
        else               vtb [(size_t)go * SEQ + gm] = h;   // transposed V
      }
    }
  }
}

// ---------------- Kernel 3: causal attention, 16 query rows per block ------
// LDS: S[16][SEQ] fp32 (256 KB) + Qs[16][DIM] bf16 (32 KB) = 288 KB < 320 KB.
__global__ __launch_bounds__(256) void attn_kernel(
    const unsigned short* __restrict__ qb,
    const unsigned short* __restrict__ kbuf,
    const unsigned short* __restrict__ vtb,
    float* __restrict__ out) {
  extern __shared__ float S[];               // [16][SEQ] fp32
  unsigned short* qs = (unsigned short*)(S + 16 * SEQ);   // [16][DIM] bf16
  __shared__ float part[16][17];
  __shared__ float row_m[16];
  __shared__ float row_l[16];

  const int ib     = blockIdx.x;             // query block [0,256)
  const int wave   = threadIdx.x >> 5;
  const int lane   = threadIdx.x & 31;
  const int nvalid = 16 * (ib + 1);          // causal key extent

  // Zero the 16-col pad block past the diagonal when nvalid % 32 == 16 so
  // the K=32 PV step reads zeros there.
  if (nvalid & 31) {
    const int rr = threadIdx.x >> 4, cc = threadIdx.x & 15;
    S[rr * SEQ + nvalid + cc] = 0.0f;
  }

  // ---- Stage the 16x1024 bf16 Q strip into LDS once (TDM if available) ---
#if __has_builtin(__builtin_amdgcn_tensor_load_to_lds)
  if (wave == 0) {
    const unsigned long long ga =
        (unsigned long long)(uintptr_t)(qb + (size_t)ib * 16 * DIM);
    const unsigned int lds_off = (unsigned int)(uintptr_t)(void*)qs;
    u32x4 g0;
    g0[0] = 1u;                                     // count=1, user descriptor
    g0[1] = lds_off;                                // lds_addr (bytes)
    g0[2] = (unsigned int)ga;                       // global_addr[31:0]
    g0[3] = (unsigned int)(ga >> 32) | (2u << 30);  // global_addr[56:32], type=2
    i32x8 g1;
    g1[0] = 0x00010000;          // wg_mask=0, data_size=1 (2 bytes)
    g1[1] = (int)(1024u << 16);  // tensor_dim0[15:0]=1024  (bits 63:48)
    g1[2] = (int)(16u << 16);    // tensor_dim0 hi=0 | tensor_dim1=16
    g1[3] = (int)(1024u << 16);  // tensor_dim1 hi=0 | tile_dim0=1024
    g1[4] = 16;                  // tile_dim1=16, tile_dim2=0
    g1[5] = 1024;                // tensor_dim0_stride[31:0] = 1024 elements
    g1[6] = 0;                   // stride hi, tensor_dim1_stride lo
    g1[7] = 0;
    const i32x4 z4 = {0, 0, 0, 0};
    const i32x8 z8 = {0, 0, 0, 0, 0, 0, 0, 0};
    // amdgpu-toolchain (clang-23) 6-arg form: (g0, g1, g2, g3, extra, cpol)
    __builtin_amdgcn_tensor_load_to_lds(g0, g1, z4, z4, z8, 0);
    __builtin_amdgcn_s_wait_tensorcnt(0);
  }
#else
  {
    const uint4* src = (const uint4*)(qb + (size_t)ib * 16 * DIM);
    uint4* dst = (uint4*)qs;
    for (int i = threadIdx.x; i < 2048; i += 256) dst[i] = src[i];
  }
#endif
  __syncthreads();

  const int n    = lane & 15;
  const int kbA  = (lane < 16) ? 0 : 8;
  const int koB  = (lane < 16) ? 0 : 8;
  const int half = (lane < 16) ? 0 : 8;

  // ---- Phase 1: S = (Q K^T) / sqrt(d), causal mask, into LDS -------------
  // Q fragments come from LDS; 4-way j-tile blocking reuses each A fragment.
  const unsigned short* aRow = qs + (size_t)n * DIM;
  int jb = wave;
  for (; jb + 24 <= ib; jb += 32) {          // full groups of 4 (strided by 8)
    const unsigned short* bR[4];
#pragma unroll
    for (int t = 0; t < 4; ++t)
      bR[t] = kbuf + (size_t)((jb + 8 * t) * 16 + n) * DIM;
    v8f acc[4] = {{}, {}, {}, {}};
    for (int kk = 0; kk < DIM; kk += 32) {
      FragB16 a;
      load_frag_a(a, aRow, kk, kbA);
#pragma unroll
      for (int t = 0; t < 4; ++t) {
        FragB16 b;
        load_frag_b(b, bR[t], kk, koB);
        acc[t] = WMMA_BF16(a, b, acc[t]);
      }
    }
#pragma unroll
    for (int t = 0; t < 4; ++t) {
      const int jbt = jb + 8 * t;
#pragma unroll
      for (int rr = 0; rr < 8; ++rr) {
        const int M = rr + half;
        const int i = ib * 16 + M;
        const int j = jbt * 16 + n;
        float s = acc[t][rr] * 0.03125f;     // 1/sqrt(1024)
        if (j > i) s = -__builtin_inff();
        S[M * SEQ + j] = s;
      }
    }
  }
  for (; jb <= ib; jb += 8) {                // remainder tiles, one at a time
    const unsigned short* bRow = kbuf + (size_t)(jb * 16 + n) * DIM;
    v8f acc = {};
    for (int kk = 0; kk < DIM; kk += 32) {
      FragB16 a, b;
      load_frag_a(a, aRow, kk, kbA);
      load_frag_b(b, bRow, kk, koB);
      acc = WMMA_BF16(a, b, acc);
    }
#pragma unroll
    for (int rr = 0; rr < 8; ++rr) {
      const int M = rr + half;
      const int i = ib * 16 + M;
      const int j = jb * 16 + n;
      float s = acc[rr] * 0.03125f;
      if (j > i) s = -__builtin_inff();
      S[M * SEQ + j] = s;
    }
  }
  __syncthreads();

  // ---- Phase 2: exact softmax stats in fp32; S <- exp(S - m) -------------
  {
    const int row = threadIdx.x & 15, sub = threadIdx.x >> 4;
    float m = -__builtin_inff();
    for (int j = sub; j < nvalid; j += 16) m = fmaxf(m, S[row * SEQ + j]);
    part[row][sub] = m;
    __syncthreads();
    if (threadIdx.x < 16) {
      float mm = part[threadIdx.x][0];
      for (int s2 = 1; s2 < 16; ++s2) mm = fmaxf(mm, part[threadIdx.x][s2]);
      row_m[threadIdx.x] = mm;
    }
    __syncthreads();
    const float mrow = row_m[row];
    float l = 0.0f;
    for (int j = sub; j < nvalid; j += 16) {
      const float e = __expf(S[row * SEQ + j] - mrow);
      S[row * SEQ + j] = e;                  // unnormalized P, fp32
      l += e;
    }
    part[row][sub] = l;
    __syncthreads();
    if (threadIdx.x < 16) {
      float ll = 0.0f;
      for (int s2 = 0; s2 < 16; ++s2) ll += part[threadIdx.x][s2];
      row_l[threadIdx.x] = ll;
    }
    __syncthreads();
  }

  // ---- Phase 3: O = P V. A from LDS (cvt->bf16), amortized over 4 B tiles.
  const int kkend = (nvalid + 31) & ~31;
#pragma unroll
  for (int og = 0; og < 2; ++og) {
    const int ot0 = wave * 8 + og * 4;       // 4 consecutive out-col tiles
    const unsigned short* bR[4];
#pragma unroll
    for (int t = 0; t < 4; ++t)
      bR[t] = vtb + (size_t)((ot0 + t) * 16 + n) * SEQ;
    v8f acc[4] = {{}, {}, {}, {}};
    for (int kk = 0; kk < kkend; kk += 32) {
      FragB16 a;
      const float* ap = S + n * SEQ + kk + kbA;
      const float4 f0 = *(const float4*)(ap);
      const float4 f1 = *(const float4*)(ap + 4);
      const float4 f2 = *(const float4*)(ap + 16);
      const float4 f3 = *(const float4*)(ap + 20);
      a.us[0]  = f2bf(f0.x); a.us[1]  = f2bf(f0.y);
      a.us[2]  = f2bf(f0.z); a.us[3]  = f2bf(f0.w);
      a.us[4]  = f2bf(f1.x); a.us[5]  = f2bf(f1.y);
      a.us[6]  = f2bf(f1.z); a.us[7]  = f2bf(f1.w);
      a.us[8]  = f2bf(f2.x); a.us[9]  = f2bf(f2.y);
      a.us[10] = f2bf(f2.z); a.us[11] = f2bf(f2.w);
      a.us[12] = f2bf(f3.x); a.us[13] = f2bf(f3.y);
      a.us[14] = f2bf(f3.z); a.us[15] = f2bf(f3.w);
#pragma unroll
      for (int t = 0; t < 4; ++t) {
        FragB16 b;
        load_frag_b(b, bR[t], kk, koB);
        acc[t] = WMMA_BF16(a, b, acc[t]);
      }
    }
#pragma unroll
    for (int t = 0; t < 4; ++t) {
#pragma unroll
      for (int rr = 0; rr < 8; ++rr) {
        const int M = rr + half;
        out[(size_t)(ib * 16 + M) * DIM + (ot0 + t) * 16 + n] =
            acc[t][rr] / row_l[M];
      }
    }
  }
}

// ---------------------------------------------------------------------------
extern "C" void kernel_launch(void* const* d_in, const int* in_sizes, int n_in,
                              void* d_out, int out_size, void* d_ws, size_t ws_size,
                              hipStream_t stream) {
  (void)in_sizes; (void)n_in; (void)out_size; (void)ws_size;
  const float* x  = (const float*)d_in[0];
  const float* wq = (const float*)d_in[1];
  const float* wk = (const float*)d_in[2];
  const float* wv = (const float*)d_in[3];
  float* out = (float*)d_out;

  char* ws = (char*)d_ws;
  // byte offsets (all 16B-aligned)
  unsigned short* xb  = (unsigned short*)(ws + 0);          //  8 MB
  unsigned short* wqb = (unsigned short*)(ws + 8388608);    //  2 MB
  unsigned short* wkb = (unsigned short*)(ws + 10485760);   //  2 MB
  unsigned short* wvb = (unsigned short*)(ws + 12582912);   //  2 MB
  unsigned short* qb  = (unsigned short*)(ws + 14680064);   //  8 MB
  unsigned short* kb  = (unsigned short*)(ws + 23068672);   //  8 MB
  unsigned short* vtb = (unsigned short*)(ws + 31457280);   //  8 MB (V^T)

  // 1) fp32 -> bf16 conversions: 7340032 elements / 256 = 28672 blocks
  cvt_bf16_kernel<<<28672, 256, 0, stream>>>(x, wq, wk, wv, xb, wqb, wkb, wvb);

  // 2) QKV projections: 3 * 128 * 32 wave-tiles, 8 waves/block -> 1536 blocks
  qkv_gemm_kernel<<<1536, 256, 0, stream>>>(xb, wqb, wkb, wvb, qb, kb, vtb);

  // 3) causal attention: 256 KB S + 32 KB staged Q of dynamic LDS
  const size_t lds_bytes = 16 * SEQ * sizeof(float) + 16 * DIM * sizeof(unsigned short);
  attn_kernel<<<256, 256, lds_bytes, stream>>>(qb, kb, vtb, out);
}
